// Attention_44409961841258
// MI455X (gfx1250) — matrix operand order repaired and checked by reading, not compiled, not run on hardware
//
#include <hip/hip_runtime.h>
#include <hip/hip_bf16.h>
#include <math.h>

typedef __attribute__((ext_vector_type(16))) _Float16 v16h;
typedef __attribute__((ext_vector_type(8)))  _Float16 h8;
typedef __attribute__((ext_vector_type(4)))  _Float16 h4;
typedef __attribute__((ext_vector_type(8)))  float    v8f;

#define CAT16(lo,hi) __builtin_shufflevector(lo,hi,0,1,2,3,4,5,6,7,8,9,10,11,12,13,14,15)

// Async global->LDS (CDNA5 VFLAT async op, tracked by ASYNCcnt).
__device__ __forceinline__ void async_b128(void* lds_dst, const void* gsrc) {
    unsigned lds_off = (unsigned)(size_t)(__attribute__((address_space(3))) void*)lds_dst;
    unsigned long long ga = (unsigned long long)gsrc;
    asm volatile("global_load_async_to_lds_b128 %0, %1, off"
                 :: "v"(lds_off), "v"(ga) : "memory");
}
__device__ __forceinline__ void wait_async0() {
#if __has_builtin(__builtin_amdgcn_s_wait_asynccnt)
    __builtin_amdgcn_s_wait_asynccnt(0);
#else
    asm volatile("s_wait_asynccnt 0" ::: "memory");
#endif
}
// wave-local ordering of LDS ops (DS pipeline is in-order within a wave)
__device__ __forceinline__ void wave_fence() {
    asm volatile("" ::: "memory");
    __builtin_amdgcn_wave_barrier();
}

// ---------------------------------------------------------------------------
// f32 -> f16 elementwise. 4 elems/thread.
// ---------------------------------------------------------------------------
__global__ __launch_bounds__(256)
void cvt16(const float* __restrict__ src, _Float16* __restrict__ dst, int n4)
{
    int i = blockIdx.x * blockDim.x + threadIdx.x;
    if (i < n4) {
        const float4 v = ((const float4*)src)[i];
        h4 o; o[0] = (_Float16)v.x; o[1] = (_Float16)v.y;
              o[2] = (_Float16)v.z; o[3] = (_Float16)v.w;
        ((h4*)dst)[i] = o;
    }
}

// ---------------------------------------------------------------------------
// WMMA GEMM:  Y = X (M x K, f16) * W16^T, W16 (N x K) row-major f16.
//   OUT_MODE : 0 = f32 row-major; 1 = f16 [b,h,l,d]; 2 = f16 [b,h,d,l] (V)
// Block = 128 thr (4 waves); block tile 128(M) x 64(N); wave tile 32 x 64.
// ---------------------------------------------------------------------------
template<int OUT_MODE>
__global__ __launch_bounds__(128)
void gemm16(const _Float16* __restrict__ X, const _Float16* __restrict__ W16,
            void* __restrict__ Yv, int M, int N, int K)
{
    const int lane  = threadIdx.x & 31;
    const int wave  = threadIdx.x >> 5;
    const int half  = lane >> 4;
    const int l16   = lane & 15;
    const int mbase = blockIdx.y * 128 + wave * 32;
    const int nbase = blockIdx.x * 64;

    v8f acc[2][4];
#pragma unroll
    for (int s = 0; s < 2; ++s)
#pragma unroll
        for (int f = 0; f < 4; ++f) acc[s][f] = v8f{};

    for (int kb = 0; kb < K; kb += 32) {
        v16h a[2];
#pragma unroll
        for (int s = 0; s < 2; ++s) {
            const _Float16* xp = X + (size_t)(mbase + s * 16 + l16) * K + kb;
            h8 lo = *(const h8*)(xp + half * 16);
            h8 hi = *(const h8*)(xp + half * 16 + 8);
            a[s] = CAT16(lo, hi);
        }
#pragma unroll
        for (int f = 0; f < 4; ++f) {
            const _Float16* wp = W16 + (size_t)(nbase + f * 16 + l16) * K + kb + half * 16;
            h8 blo = *(const h8*)wp;
            h8 bhi = *(const h8*)(wp + 8);
            v16h bf = CAT16(blo, bhi);
            acc[0][f] = __builtin_amdgcn_wmma_f32_16x16x32_f16(
                false, a[0], false, bf, (short)0, acc[0][f], false, false);
            acc[1][f] = __builtin_amdgcn_wmma_f32_16x16x32_f16(
                false, a[1], false, bf, (short)0, acc[1][f], false, false);
        }
    }

#pragma unroll
    for (int s = 0; s < 2; ++s)
#pragma unroll
        for (int f = 0; f < 4; ++f)
#pragma unroll
            for (int r = 0; r < 8; ++r) {
                const int orow = mbase + s * 16 + r + half * 8;
                const int ocol = nbase + f * 16 + l16;
                const float v = acc[s][f][r];
                if (OUT_MODE == 0) {
                    ((float*)Yv)[(size_t)orow * N + ocol] = v;
                } else {
                    _Float16* Y = (_Float16*)Yv;
                    const int bb = orow >> 11, l = orow & 2047;
                    const int hh = ocol >> 7,  dd = ocol & 127;
                    if (OUT_MODE == 1)
                        Y[(((size_t)bb * 16 + hh) * 2048 + l) * 128 + dd] = (_Float16)v;
                    else
                        Y[(((size_t)bb * 16 + hh) * 128 + dd) * 2048 + l] = (_Float16)v;
                }
            }
}

// ---------------------------------------------------------------------------
// Flash attention. Qh/Kh: [bh, L, 128] f16; VhT: [bh, 128, L] f16 (d-major).
// Block = 256 thr (8 waves) -> 128 q rows per (b,h); 32-key tiles, async
// double-buffered in LDS. B-fragment LDS loads ping-pong between two
// register sets so WMMAs only partially wait on DScnt.
// Output Oattn: [b, l, h*128+d] f16.
// ---------------------------------------------------------------------------
__global__ __launch_bounds__(256)
void fattn(const _Float16* __restrict__ Qh, const _Float16* __restrict__ Kh,
           const _Float16* __restrict__ VhT, const unsigned char* __restrict__ mask,
           _Float16* __restrict__ Oattn)
{
    constexpr int L = 2048, D = 128, NT = L / 32;
    __shared__ __align__(16) _Float16 Kt[2][32 * 128];   // [buf][key][d]
    __shared__ __align__(16) _Float16 Vt[2][128 * 32];   // [buf][d][key]
    __shared__ __align__(16) _Float16 Ps[8][16 * 32];    // per-wave P staging

    const int lane  = threadIdx.x & 31;
    const int wave  = threadIdx.x >> 5;
    const int half  = lane >> 4;
    const int l16   = lane & 15;
    const int bh    = blockIdx.y;            // b*16 + h
    const int b     = bh >> 4;
    const int h     = bh & 15;
    const int qbase = blockIdx.x * 128 + wave * 16;

    const _Float16* Kbase = Kh  + (size_t)bh * L * D;
    const _Float16* Vbase = VhT + (size_t)bh * D * L; // d-major

    // ---- Q fragments, resident for the whole kernel ----
    v16h qa[4];
    {
        const _Float16* qrow = Qh + ((size_t)bh * L + qbase + l16) * D;
#pragma unroll
        for (int f = 0; f < 4; ++f) {
            h8 lo = *(const h8*)(qrow + f * 32 + half * 16);
            h8 hi = *(const h8*)(qrow + f * 32 + half * 16 + 8);
            qa[f] = CAT16(lo, hi);
        }
    }

    float m_s[8], l_s[8];
#pragma unroll
    for (int r = 0; r < 8; ++r) { m_s[r] = -1e30f; l_s[r] = 0.0f; }
    v8f o[8];
#pragma unroll
    for (int n = 0; n < 8; ++n) o[n] = v8f{};

    const float scale = 0.08838834764831845f; // 1/sqrt(128)

    // cooperative async tile fetch: 256 threads, 16 halfs (2x b128) per array
    auto load_tile = [&](int kb, int bufi) {
        const int key = threadIdx.x & 31;
        const int dc  = (threadIdx.x >> 5) * 16;          // 0..112
        const _Float16* ks = Kbase + (size_t)(kb + key) * D + dc;
        async_b128(&Kt[bufi][key * 128 + dc],     ks);
        async_b128(&Kt[bufi][key * 128 + dc + 8], ks + 8);
        const int d  = threadIdx.x >> 1;                   // 0..127
        const int c2 = (threadIdx.x & 1) * 16;             // 0/16
        const _Float16* vs = Vbase + (size_t)d * L + kb + c2;
        async_b128(&Vt[bufi][d * 32 + c2],     vs);
        async_b128(&Vt[bufi][d * 32 + c2 + 8], vs + 8);
    };

    load_tile(0, 0);
    wait_async0();
    __syncthreads();

    for (int t = 0; t < NT; ++t) {
        const int cur = t & 1;
        const int kb  = t * 32;
        if (t + 1 < NT) load_tile(kb + 32, cur ^ 1);   // prefetch next tile

        // ---- S = Q * K^T : 8 WMMAs; B-frag loads ping-pong across 2 reg sets
        //      so each WMMA only waits on its own fragment's loads ----
        v8f sacc[2] = {v8f{}, v8f{}};
        {
            auto kaddr = [&](int i) {   // i = f*2 + sub
                const int sub = i & 1, f = i >> 1;
                return (const _Float16*)&Kt[cur][(sub * 16 + l16) * 128 + f * 32 + half * 16];
            };
            h8 blo[2], bhi[2];
            blo[0] = *(const h8*)kaddr(0);
            bhi[0] = *(const h8*)(kaddr(0) + 8);
#pragma unroll
            for (int i = 0; i < 8; ++i) {
                const int cb = i & 1;
                if (i < 7) {                       // next frag -> other reg set
                    const _Float16* np = kaddr(i + 1);
                    blo[cb ^ 1] = *(const h8*)np;
                    bhi[cb ^ 1] = *(const h8*)(np + 8);
                }
                v16h bf = CAT16(blo[cb], bhi[cb]);
                sacc[i & 1] = __builtin_amdgcn_wmma_f32_16x16x32_f16(
                    false, qa[i >> 1], false, bf, (short)0, sacc[i & 1], false, false);
            }
        }

        // ---- scale + mask + online softmax ----
        float p0[8], p1[8], mt[8];
#pragma unroll
        for (int r = 0; r < 8; ++r) {
            const int qrow = qbase + r + half * 8;
            const size_t mrow = ((size_t)b * L + qrow) * L + kb;
            const unsigned char mk0 = mask[mrow + l16];
            const unsigned char mk1 = mask[mrow + 16 + l16];
            p0[r] = mk0 ? sacc[0][r] * scale : -1e30f;
            p1[r] = mk1 ? sacc[1][r] * scale : -1e30f;
            mt[r] = fmaxf(p0[r], p1[r]);
        }
#pragma unroll
        for (int x = 1; x < 16; x <<= 1)
#pragma unroll
            for (int r = 0; r < 8; ++r)
                mt[r] = fmaxf(mt[r], __shfl_xor(mt[r], x, 32));

        float alpha[8], rsum[8];
#pragma unroll
        for (int r = 0; r < 8; ++r) {
            const float mn = fmaxf(m_s[r], mt[r]);
            alpha[r] = expf(m_s[r] - mn);
            p0[r] = expf(p0[r] - mn);
            p1[r] = expf(p1[r] - mn);
            rsum[r] = p0[r] + p1[r];
            m_s[r] = mn;
        }
#pragma unroll
        for (int x = 1; x < 16; x <<= 1)
#pragma unroll
            for (int r = 0; r < 8; ++r)
                rsum[r] += __shfl_xor(rsum[r], x, 32);
#pragma unroll
        for (int r = 0; r < 8; ++r)
            l_s[r] = l_s[r] * alpha[r] + rsum[r];
#pragma unroll
        for (int n = 0; n < 8; ++n)
#pragma unroll
            for (int r = 0; r < 8; ++r)
                o[n][r] *= alpha[r];

        // ---- stage P (C layout -> LDS, wave-private region) ----
        {
            _Float16* pw = &Ps[wave][0];
#pragma unroll
            for (int r = 0; r < 8; ++r) {
                pw[(r + half * 8) * 32 + l16]      = (_Float16)p0[r];
                pw[(r + half * 8) * 32 + 16 + l16] = (_Float16)p1[r];
            }
        }
        wave_fence();
        v16h pa;
        {
            const _Float16* pr = &Ps[wave][0] + l16 * 32;
            h8 lo = *(const h8*)(pr + half * 16);
            h8 hi = *(const h8*)(pr + half * 16 + 8);
            pa = CAT16(lo, hi);
        }

        // ---- O += P * V : 8 WMMAs, same A operand, ping-pong loads ----
        {
            auto vaddr = [&](int n) {
                return (const _Float16*)&Vt[cur][(n * 16 + l16) * 32 + half * 16];
            };
            h8 blo[2], bhi[2];
            blo[0] = *(const h8*)vaddr(0);
            bhi[0] = *(const h8*)(vaddr(0) + 8);
#pragma unroll
            for (int n = 0; n < 8; ++n) {
                const int cb = n & 1;
                if (n < 7) {
                    const _Float16* np = vaddr(n + 1);
                    blo[cb ^ 1] = *(const h8*)np;
                    bhi[cb ^ 1] = *(const h8*)(np + 8);
                }
                v16h bf = CAT16(blo[cb], bhi[cb]);
                o[n] = __builtin_amdgcn_wmma_f32_16x16x32_f16(
                    false, pa, false, bf, (short)0, o[n], false, false);
            }
        }

        wait_async0();    // next tile's async LDS writes done
        __syncthreads();  // all waves done reading cur before it is reused
    }

    // ---- epilogue: normalize and store [b, l, h*128+d] ----
    float inv[8];
#pragma unroll
    for (int r = 0; r < 8; ++r)
        inv[r] = (l_s[r] > 0.0f) ? (1.0f / l_s[r]) : 0.0f;
#pragma unroll
    for (int n = 0; n < 8; ++n)
#pragma unroll
        for (int r = 0; r < 8; ++r) {
            const int orow = qbase + r + half * 8;
            const int ocol = h * 128 + n * 16 + l16;
            Oattn[((size_t)b * L + orow) * 2048 + ocol] = (_Float16)(o[n][r] * inv[r]);
        }
}

// ---------------------------------------------------------------------------
extern "C" void kernel_launch(void* const* d_in, const int* in_sizes, int n_in,
                              void* d_out, int out_size, void* d_ws, size_t ws_size,
                              hipStream_t stream)
{
    const float*         query = (const float*)d_in[0];
    const float*         kv    = (const float*)d_in[1];
    const unsigned char* mask  = (const unsigned char*)d_in[2];
    const float*         Wq    = (const float*)d_in[3];
    const float*         Wk    = (const float*)d_in[4];
    const float*         Wv    = (const float*)d_in[5];
    const float*         Wo    = (const float*)d_in[6];
    float*               out   = (float*)d_out;

    // ws: Qh|Kh|VhT|Oattn (16 MB each) | Wq16|Wk16|Wv16|Wo16 (8 MB) | X16 (16 MB)
    _Float16* ws = (_Float16*)d_ws;
    const size_t ACT = (size_t)2 * 16 * 2048 * 128;  // 8388608 halfs
    const size_t WSZ = (size_t)2048 * 2048;          // 4194304 halfs
    _Float16* Qh   = ws;
    _Float16* Kh   = ws + ACT;
    _Float16* VhT  = ws + 2 * ACT;
    _Float16* Oa   = ws + 3 * ACT;
    _Float16* Wq16 = ws + 4 * ACT;
    _Float16* Wk16 = Wq16 + WSZ;
    _Float16* Wv16 = Wk16 + WSZ;
    _Float16* Wo16 = Wv16 + WSZ;
    _Float16* X16  = Wo16 + WSZ;   // staged f16 activations (reused)

    const int wn4 = (int)(WSZ / 4), wcb = (wn4 + 255) / 256;
    const int xn4 = (int)(ACT / 4), xcb = (xn4 + 255) / 256;
    cvt16<<<wcb, 256, 0, stream>>>(Wq, Wq16, wn4);
    cvt16<<<wcb, 256, 0, stream>>>(Wk, Wk16, wn4);
    cvt16<<<wcb, 256, 0, stream>>>(Wv, Wv16, wn4);
    cvt16<<<wcb, 256, 0, stream>>>(Wo, Wo16, wn4);

    const dim3 gg(2048 / 64, 4096 / 128); // (N tiles, M tiles)
    cvt16<<<xcb, 256, 0, stream>>>(query, X16, xn4);
    gemm16<1><<<gg, 128, 0, stream>>>(X16, Wq16, Qh,  4096, 2048, 2048);
    cvt16<<<xcb, 256, 0, stream>>>(kv, X16, xn4);
    gemm16<1><<<gg, 128, 0, stream>>>(X16, Wk16, Kh,  4096, 2048, 2048);
    gemm16<2><<<gg, 128, 0, stream>>>(X16, Wv16, VhT, 4096, 2048, 2048);
    fattn<<<dim3(2048 / 128, 32), 256, 0, stream>>>(Qh, Kh, VhT, mask, Oa);
    gemm16<0><<<gg, 128, 0, stream>>>(Oa, Wo16, out, 4096, 2048, 2048);
}
